// SwinTransformerBlock_67199058314021
// MI455X (gfx1250) — matrix operand order repaired and checked by reading, not compiled, not run on hardware
//
#include <hip/hip_runtime.h>

// ---------------------------------------------------------------------------
// Swin Transformer block for MI455X (gfx1250, wave32, WMMA f32_16x16x32_f16)
//   kernel 1: pack weights fp32 -> f16 in WMMA B-fragment layout (coalesced)
//   kernel 2: per-window fused LN1 + shift + QKV + attention + proj + residual
//   kernel 3: per-128-token fused LN2 + FC1 + GELU + FC2 + residual
// ---------------------------------------------------------------------------

typedef __attribute__((ext_vector_type(16))) _Float16 v16h;
typedef __attribute__((ext_vector_type(8)))  _Float16 v8h;
typedef __attribute__((ext_vector_type(4)))  _Float16 v4h;
typedef __attribute__((ext_vector_type(8)))  float    v8f;

#define DEV __device__ __forceinline__

constexpr int DIMc  = 128;
constexpr int HEADS = 4;
constexpr int WHc = 4, WWc = 32;      // window 4x32 -> N = 128 tokens
constexpr int Hc = 64, Wc = 512, Bc = 8;
constexpr int SHc = 2, SWc = 2;
constexpr int NWWc = Wc / WWc;        // 16
constexpr int NHWc = Hc / WHc;        // 16
constexpr float SCALEc = 0.17677669529663687f;   // 32^-0.5

// LDS strides (halves / floats), padded against bank conflicts
constexpr int XS_LD = 136;   // X / P(probs) / O / H(mlp) tiles, 128 cols
constexpr int QK_LD = 40;    // Q,K per-head [128][32]
constexpr int VT_LD = 136;   // V^T per-head [32][128]
constexpr int S_LD  = 132;   // score rows (f32)
constexpr int G_LD  = 520;   // MLP hidden [128][512]

DEV v8f wmma16(v16h a, v16h b, v8f c) {
    return __builtin_amdgcn_wmma_f32_16x16x32_f16(
        false, a, false, b, (short)0, c, false, false);
}
DEV v8f zero8() {
    v8f z;
#pragma unroll
    for (int i = 0; i < 8; ++i) z[i] = 0.f;
    return z;
}
DEV v16h hcat(v8h lo, v8h hi) {
    return __builtin_shufflevector(lo, hi, 0,1,2,3,4,5,6,7,8,9,10,11,12,13,14,15);
}

// A fragment (16Mx32K, f16): lane L holds row m0+(L&15); k = (L>>4)*16 + e
// -> one contiguous 16-half read.
DEV v16h load_A(const _Float16* base, int ld, int m0, int k0) {
    const int lane = threadIdx.x & 31;
    const _Float16* p = base + (m0 + (lane & 15)) * ld + k0 + ((lane >> 4) << 4);
    return hcat(*(const v8h*)p, *(const v8h*)(p + 8));
}
// B fragment (32Kx16N) where B[k][n] is stored as buf[n*ld + k] (column-major
// source, e.g. K^T from K[j][d] or V from V^T[d][j]): lane L holds col
// n0+(L&15); k = (L>>4)*16 + e -> one contiguous 16-half read.
DEV v16h load_Bt(const _Float16* base, int ld, int k0, int n0) {
    const int lane = threadIdx.x & 31;
    const _Float16* p = base + (n0 + (lane & 15)) * ld + k0 + ((lane >> 4) << 4);
    return hcat(*(const v8h*)p, *(const v8h*)(p + 8));
}
// B fragment from pre-packed global weights: fragment (nt*KT+kt) is 512 halves,
// lane-major -> each lane reads 32 contiguous bytes (fully coalesced).
DEV v16h load_Bg(const _Float16* pack, int KT, int nt, int kt) {
    const int lane = threadIdx.x & 31;
    const _Float16* p = pack + (((size_t)(nt * KT + kt)) << 9) + (lane << 4);
    return hcat(*(const v8h*)p, *(const v8h*)(p + 8));
}

DEV float wsum(float v) {
#pragma unroll
    for (int m = 16; m; m >>= 1) v += __shfl_xor(v, m, 32);
    return v;
}
DEV float wmax(float v) {
#pragma unroll
    for (int m = 16; m; m >>= 1) v = fmaxf(v, __shfl_xor(v, m, 32));
    return v;
}

// --------------------------------------------------------------------------
// Weight packer: W[K][N] fp32 -> f16 fragments matching load_Bg layout.
// --------------------------------------------------------------------------
__global__ void swin_pack_b(const float* __restrict__ Wsrc,
                            _Float16* __restrict__ dst, int K, int N) {
    int idx = blockIdx.x * 256 + threadIdx.x;
    if (idx >= K * N) return;
    int e    = idx & 15;
    int lane = (idx >> 4) & 31;
    int frag = idx >> 9;
    int KT   = K >> 5;
    int kt   = frag % KT;
    int nt   = frag / KT;
    int k = kt * 32 + ((lane >> 4) << 4) + e;
    int n = nt * 16 + (lane & 15);
    dst[idx] = (_Float16)Wsrc[k * N + n];
}

// --------------------------------------------------------------------------
// Kernel 2: one block per window (B*NWIN = 2048 blocks, 256 threads = 8 waves)
// --------------------------------------------------------------------------
__global__ __launch_bounds__(256) void swin_attn_kernel(
    const float* __restrict__ x,  const float* __restrict__ g1,
    const float* __restrict__ b1, const float* __restrict__ qkv_b,
    const float* __restrict__ rpb, const float* __restrict__ proj_b,
    const _Float16* __restrict__ qkv_pack,
    const _Float16* __restrict__ proj_pack,
    float* __restrict__ out) {
    extern __shared__ char smem[];
    _Float16* Qs = (_Float16*)smem;                 // [4][128][40]
    _Float16* Ks = Qs + HEADS * 128 * QK_LD;        // [4][128][40]
    _Float16* Vt = Ks + HEADS * 128 * QK_LD;        // [4][32][136]
    _Float16* Xs = Vt + HEADS * 32 * VT_LD;         // [128][136] (also P f16)
    float*    Sb = (float*)(Xs + 128 * XS_LD);      // [128][132] scores f32
    _Float16* Os = (_Float16*)(Sb + 128 * S_LD);    // [128][136] attn out f16
    float*    rp = (float*)(Os + 128 * XS_LD);      // 441*4 rel-pos bias
    int*      rg = (int*)(rp + 441 * HEADS);        // 128 region ids

    const int tid  = threadIdx.x;
    const int lane = tid & 31;
    const int wv   = tid >> 5;
    const int wid  = blockIdx.x;
    const int b    = wid >> 8;         // / 256 windows per batch
    const int wr   = (wid >> 4) & 15;
    const int wc   = wid & 15;

    // ---- phase 0: region ids, rpb staging, LN1 over shifted window tokens
    if (tid < 128) {
        int h = wr * WHc + (tid >> 5), w = wc * WWc + (tid & 31);
        int rh = (h < Hc - WHc) ? 0 : ((h < Hc - SHc) ? 1 : 2);
        int rw = (w < Wc - WWc) ? 0 : ((w < Wc - SWc) ? 1 : 2);
        rg[tid] = rh * 3 + rw;
    }
    for (int i = tid; i < 441 * HEADS; i += 256) rp[i] = rpb[i];

#pragma unroll 1
    for (int i = 0; i < 16; ++i) {
        int t  = wv * 16 + i;
        int sh = (wr * WHc + (t >> 5) + SHc) & (Hc - 1);
        int sw = (wc * WWc + (t & 31) + SWc) & (Wc - 1);
        size_t g = (size_t)(b * Hc + sh) * Wc + sw;
        float4 v = ((const float4*)(x + g * DIMc))[lane];
        float mean = wsum(v.x + v.y + v.z + v.w) * (1.f / DIMc);
        float sq   = wsum(v.x*v.x + v.y*v.y + v.z*v.z + v.w*v.w) * (1.f / DIMc);
        float rstd = rsqrtf(sq - mean * mean + 1e-5f);
        int c = lane * 4;
        v4h o;
        o[0] = (_Float16)((v.x - mean) * rstd * g1[c+0] + b1[c+0]);
        o[1] = (_Float16)((v.y - mean) * rstd * g1[c+1] + b1[c+1]);
        o[2] = (_Float16)((v.z - mean) * rstd * g1[c+2] + b1[c+2]);
        o[3] = (_Float16)((v.w - mean) * rstd * g1[c+3] + b1[c+3]);
        *(v4h*)(Xs + t * XS_LD + c) = o;
    }
    __syncthreads();

    // ---- phase 1: QKV = Xs(128x128) * Wqkv(128x384), scatter to Q/K/V^T
    {
        v16h a0 = load_A(Xs, XS_LD, wv * 16, 0);
        v16h a1 = load_A(Xs, XS_LD, wv * 16, 32);
        v16h a2 = load_A(Xs, XS_LD, wv * 16, 64);
        v16h a3 = load_A(Xs, XS_LD, wv * 16, 96);
#pragma unroll 1
        for (int nt = 0; nt < 24; ++nt) {
            v8f acc = zero8();
            acc = wmma16(a0, load_Bg(qkv_pack, 4, nt, 0), acc);
            acc = wmma16(a1, load_Bg(qkv_pack, 4, nt, 1), acc);
            acc = wmma16(a2, load_Bg(qkv_pack, 4, nt, 2), acc);
            acc = wmma16(a3, load_Bg(qkv_pack, 4, nt, 3), acc);
            int col  = nt * 16 + (lane & 15);
            float bb = qkv_b[col];
            int mb   = wv * 16 + ((lane >> 4) << 3);
            if (col < 128) {                               // Q (pre-scaled)
                int hd = col >> 5, d = col & 31;
                _Float16* q = Qs + hd * 128 * QK_LD + d;
#pragma unroll
                for (int r = 0; r < 8; ++r)
                    q[(mb + r) * QK_LD] = (_Float16)((acc[r] + bb) * SCALEc);
            } else if (col < 256) {                        // K
                int cc = col - 128, hd = cc >> 5, d = cc & 31;
                _Float16* k = Ks + hd * 128 * QK_LD + d;
#pragma unroll
                for (int r = 0; r < 8; ++r)
                    k[(mb + r) * QK_LD] = (_Float16)(acc[r] + bb);
            } else {                                       // V transposed
                int cc = col - 256, hd = cc >> 5, d = cc & 31;
                _Float16* vv = Vt + hd * 32 * VT_LD + d * VT_LD;
#pragma unroll
                for (int r = 0; r < 8; ++r)
                    vv[mb + r] = (_Float16)(acc[r] + bb);
            }
        }
    }
    __syncthreads();

    // ---- phase 2: per-head attention (heads sequential; 8 waves cooperate)
#pragma unroll 1
    for (int hh = 0; hh < HEADS; ++hh) {
        const _Float16* Qh = Qs + hh * 128 * QK_LD;
        const _Float16* Kh = Ks + hh * 128 * QK_LD;
        const _Float16* Vh = Vt + hh * 32 * VT_LD;
        // S = Q K^T (+ rel-pos bias + shift mask), wave wv owns M-tile row wv
        v16h aq = load_A(Qh, QK_LD, wv * 16, 0);
#pragma unroll 1
        for (int nt = 0; nt < 8; ++nt) {
            v8f s = wmma16(aq, load_Bt(Kh, QK_LD, 0, nt * 16), zero8());
            int j  = nt * 16 + (lane & 15);
            int ib = wv * 16 + ((lane >> 4) << 3);
            int rj = rg[j];
#pragma unroll
            for (int r = 0; r < 8; ++r) {
                int i  = ib + r;
                int dh = (i >> 5) - (j >> 5);
                int dw = (i & 31) - (j & 31);
                float bias = rp[((dh + 3) * 63 + dw + 31) * HEADS + hh];
                float mv   = (rg[i] != rj) ? -100.f : 0.f;
                Sb[i * S_LD + j] = s[r] + bias + mv;
            }
        }
        __syncthreads();
        // softmax rows [wv*16, wv*16+16), probs -> Xs as f16
#pragma unroll 1
        for (int rr = 0; rr < 16; ++rr) {
            int row = wv * 16 + rr;
            float v0 = Sb[row * S_LD + lane];
            float v1 = Sb[row * S_LD + lane + 32];
            float v2 = Sb[row * S_LD + lane + 64];
            float v3 = Sb[row * S_LD + lane + 96];
            float mx = wmax(fmaxf(fmaxf(v0, v1), fmaxf(v2, v3)));
            v0 = __expf(v0 - mx); v1 = __expf(v1 - mx);
            v2 = __expf(v2 - mx); v3 = __expf(v3 - mx);
            float inv = 1.f / wsum(v0 + v1 + v2 + v3);
            Xs[row * XS_LD + lane     ] = (_Float16)(v0 * inv);
            Xs[row * XS_LD + lane + 32] = (_Float16)(v1 * inv);
            Xs[row * XS_LD + lane + 64] = (_Float16)(v2 * inv);
            Xs[row * XS_LD + lane + 96] = (_Float16)(v3 * inv);
        }
        __syncthreads();
        // O_h = P(128x128) * V_h(128x32) -> columns hh*32.. of Os
#pragma unroll 1
        for (int nt = 0; nt < 2; ++nt) {
            v8f acc = zero8();
#pragma unroll
            for (int kt = 0; kt < 4; ++kt)
                acc = wmma16(load_A(Xs, XS_LD, wv * 16, kt * 32),
                             load_Bt(Vh, VT_LD, kt * 32, nt * 16), acc);
            int n  = hh * 32 + nt * 16 + (lane & 15);
            int mb = wv * 16 + ((lane >> 4) << 3);
#pragma unroll
            for (int r = 0; r < 8; ++r)
                Os[(mb + r) * XS_LD + n] = (_Float16)acc[r];
        }
        __syncthreads();
    }

    // ---- phase 3: proj + residual; out = x + o at shift-reversed position
    {
        v16h a0 = load_A(Os, XS_LD, wv * 16, 0);
        v16h a1 = load_A(Os, XS_LD, wv * 16, 32);
        v16h a2 = load_A(Os, XS_LD, wv * 16, 64);
        v16h a3 = load_A(Os, XS_LD, wv * 16, 96);
#pragma unroll 1
        for (int nt = 0; nt < 8; ++nt) {
            v8f acc = zero8();
            acc = wmma16(a0, load_Bg(proj_pack, 4, nt, 0), acc);
            acc = wmma16(a1, load_Bg(proj_pack, 4, nt, 1), acc);
            acc = wmma16(a2, load_Bg(proj_pack, 4, nt, 2), acc);
            acc = wmma16(a3, load_Bg(proj_pack, 4, nt, 3), acc);
            int c    = nt * 16 + (lane & 15);
            float pb = proj_b[c];
            int mb   = wv * 16 + ((lane >> 4) << 3);
#pragma unroll
            for (int r = 0; r < 8; ++r) {
                int t  = mb + r;
                int sh = (wr * WHc + (t >> 5) + SHc) & (Hc - 1);
                int sw = (wc * WWc + (t & 31) + SWc) & (Wc - 1);
                size_t g = ((size_t)(b * Hc + sh) * Wc + sw) * DIMc + c;
                out[g] = x[g] + acc[r] + pb;
            }
        }
    }
}

// --------------------------------------------------------------------------
// Kernel 3: MLP, one block per 128 tokens (2048 blocks, 256 threads)
// reads r = x+o from `out`, writes out = r + fc2(gelu(fc1(ln2(r))))
// --------------------------------------------------------------------------
__global__ __launch_bounds__(256) void swin_mlp_kernel(
    const float* __restrict__ g2,  const float* __restrict__ b2,
    const float* __restrict__ fc1_b, const float* __restrict__ fc2_b,
    const _Float16* __restrict__ fc1_pack,
    const _Float16* __restrict__ fc2_pack,
    float* __restrict__ out) {
    extern __shared__ char smem[];
    _Float16* Hs = (_Float16*)smem;        // [128][136] ln2 activations
    _Float16* Gs = Hs + 128 * XS_LD;       // [128][520] gelu(fc1) hidden

    const int tid  = threadIdx.x;
    const int lane = tid & 31;
    const int wv   = tid >> 5;
    const size_t t0 = (size_t)blockIdx.x * 128;

    // LN2
#pragma unroll 1
    for (int i = 0; i < 16; ++i) {
        int t = wv * 16 + i;
        float4 v = ((const float4*)(out + (t0 + t) * DIMc))[lane];
        float mean = wsum(v.x + v.y + v.z + v.w) * (1.f / DIMc);
        float sq   = wsum(v.x*v.x + v.y*v.y + v.z*v.z + v.w*v.w) * (1.f / DIMc);
        float rstd = rsqrtf(sq - mean * mean + 1e-5f);
        int c = lane * 4;
        v4h o;
        o[0] = (_Float16)((v.x - mean) * rstd * g2[c+0] + b2[c+0]);
        o[1] = (_Float16)((v.y - mean) * rstd * g2[c+1] + b2[c+1]);
        o[2] = (_Float16)((v.z - mean) * rstd * g2[c+2] + b2[c+2]);
        o[3] = (_Float16)((v.w - mean) * rstd * g2[c+3] + b2[c+3]);
        *(v4h*)(Hs + t * XS_LD + c) = o;
    }
    __syncthreads();

    // FC1 (128 -> 512) + exact GELU
    {
        v16h a0 = load_A(Hs, XS_LD, wv * 16, 0);
        v16h a1 = load_A(Hs, XS_LD, wv * 16, 32);
        v16h a2 = load_A(Hs, XS_LD, wv * 16, 64);
        v16h a3 = load_A(Hs, XS_LD, wv * 16, 96);
#pragma unroll 1
        for (int nt = 0; nt < 32; ++nt) {
            v8f acc = zero8();
            acc = wmma16(a0, load_Bg(fc1_pack, 4, nt, 0), acc);
            acc = wmma16(a1, load_Bg(fc1_pack, 4, nt, 1), acc);
            acc = wmma16(a2, load_Bg(fc1_pack, 4, nt, 2), acc);
            acc = wmma16(a3, load_Bg(fc1_pack, 4, nt, 3), acc);
            int n    = nt * 16 + (lane & 15);
            float bb = fc1_b[n];
            int mb   = wv * 16 + ((lane >> 4) << 3);
#pragma unroll
            for (int r = 0; r < 8; ++r) {
                float u = acc[r] + bb;
                float gel = 0.5f * u * (1.f + erff(u * 0.70710678118654752f));
                Gs[(mb + r) * G_LD + n] = (_Float16)gel;
            }
        }
    }
    __syncthreads();

    // FC2 (512 -> 128) + residual, written straight to out
    {
        v8f acc[8];
#pragma unroll
        for (int nt = 0; nt < 8; ++nt) acc[nt] = zero8();
#pragma unroll 1
        for (int kt = 0; kt < 16; ++kt) {
            v16h a = load_A(Gs, G_LD, wv * 16, kt * 32);
#pragma unroll
            for (int nt = 0; nt < 8; ++nt)
                acc[nt] = wmma16(a, load_Bg(fc2_pack, 16, nt, kt), acc[nt]);
        }
        int mb = wv * 16 + ((lane >> 4) << 3);
#pragma unroll 1
        for (int nt = 0; nt < 8; ++nt) {
            int n    = nt * 16 + (lane & 15);
            float bb = fc2_b[n];
#pragma unroll
            for (int r = 0; r < 8; ++r) {
                size_t g = (t0 + mb + r) * DIMc + n;
                out[g] = out[g] + acc[nt][r] + bb;
            }
        }
    }
}

// --------------------------------------------------------------------------
extern "C" void kernel_launch(void* const* d_in, const int* in_sizes, int n_in,
                              void* d_out, int out_size, void* d_ws, size_t ws_size,
                              hipStream_t stream) {
    const float* x      = (const float*)d_in[0];
    const float* g1     = (const float*)d_in[1];
    const float* b1     = (const float*)d_in[2];
    const float* qkv_w  = (const float*)d_in[3];
    const float* qkv_b  = (const float*)d_in[4];
    const float* rpb    = (const float*)d_in[5];
    const float* proj_w = (const float*)d_in[6];
    const float* proj_b = (const float*)d_in[7];
    const float* g2     = (const float*)d_in[8];
    const float* b2     = (const float*)d_in[9];
    const float* fc1_w  = (const float*)d_in[10];
    const float* fc1_b  = (const float*)d_in[11];
    const float* fc2_w  = (const float*)d_in[12];
    const float* fc2_b  = (const float*)d_in[13];
    float* out = (float*)d_out;

    _Float16* ws        = (_Float16*)d_ws;
    _Float16* qkv_pack  = ws;                          // 128*384 halves
    _Float16* proj_pack = qkv_pack + 128 * 384;        // 128*128
    _Float16* fc1_pack  = proj_pack + 128 * 128;       // 128*512
    _Float16* fc2_pack  = fc1_pack + 128 * 512;        // 512*128

    swin_pack_b<<<(128 * 384 + 255) / 256, 256, 0, stream>>>(qkv_w,  qkv_pack, 128, 384);
    swin_pack_b<<<(128 * 128 + 255) / 256, 256, 0, stream>>>(proj_w, proj_pack, 128, 128);
    swin_pack_b<<<(128 * 512 + 255) / 256, 256, 0, stream>>>(fc1_w,  fc1_pack, 128, 512);
    swin_pack_b<<<(512 * 128 + 255) / 256, 256, 0, stream>>>(fc2_w,  fc2_pack, 512, 128);

    // dynamic LDS: Q+K (2*4*128*40) + Vt (32*136*4) + Xs (128*136) halves,
    // + Sb (128*132 f32) + Os (128*136 halves) + rpb (441*4 f32) + rg (128 i32)
    size_t smem1 = (size_t)(HEADS * 128 * QK_LD * 2 + HEADS * 32 * VT_LD
                            + 128 * XS_LD) * sizeof(_Float16)
                 + (size_t)(128 * S_LD) * sizeof(float)
                 + (size_t)(128 * XS_LD) * sizeof(_Float16)
                 + (size_t)(441 * HEADS) * sizeof(float)
                 + 128 * sizeof(int);                   // ~261.5 KB (<320 KB/WGP)
    swin_attn_kernel<<<Bc * NHWc * NWWc, 256, smem1, stream>>>(
        x, g1, b1, qkv_b, rpb, proj_b, qkv_pack, proj_pack, out);

    size_t smem2 = (size_t)(128 * XS_LD + 128 * G_LD) * sizeof(_Float16); // ~164 KB
    swin_mlp_kernel<<<(Bc * Hc * Wc) / 128, 256, smem2, stream>>>(
        g2, b2, fc1_b, fc2_b, fc1_pack, fc2_pack, out);
}
